// GraphAtteLayer_77043123356298
// MI455X (gfx1250) — matrix operand order repaired and checked by reading, not compiled, not run on hardware
//
#include <hip/hip_runtime.h>
#include <math.h>

// GAT layer (3 fields): h = x@W (WMMA f16->f32), edge softmax over dst,
// weighted scatter, head-mean + bias + leaky_relu(0.01).
// Workspace use: N*(128 + 8*4 + 128) floats = 115.2 MB.

#define DIN 32
#define HC  128
#define NH  8
#define CC  16

typedef __attribute__((ext_vector_type(16))) _Float16 v16h;
typedef __attribute__((ext_vector_type(8)))  float    v8f;

// ---------------- Kernel 1: h = f16(x) @ f16(W) via v_wmma_f32_16x16x32_f16
// Block = 256 threads = 8 waves; each wave owns a 16-row M tile and sweeps the
// 8 column tiles (128 cols). W staged transposed in LDS as f16.
// The bounds predicate is BLOCK-uniform (pure SGPR), so 781/782 blocks take a
// scalar-branch fast path with no exec masking at all.
__global__ __launch_bounds__(256) void gat_gemm_wmma(
    const float* __restrict__ x, const float* __restrict__ W,
    float* __restrict__ h, int N)
{
    __shared__ _Float16 WhT[HC * DIN];  // [n][k] halves, 8 KB
    for (int i = threadIdx.x; i < DIN * HC; i += 256) {
        int k = i >> 7, n = i & 127;    // W is [k][n] row-major
        WhT[n * DIN + k] = (_Float16)W[i];
    }
    __syncthreads();

    const int  lane  = threadIdx.x & 31;
    const int  wave  = threadIdx.x >> 5;
    const int  mbase = blockIdx.x * 128 + wave * 16;
    const int  mrow  = mbase + (lane & 15);
    const int  kbase = (lane < 16) ? 0 : 8;        // A-frag K base (16-bit A layout)
    const bool blockFull = ((blockIdx.x + 1) * 128 <= N);  // scalar (SGPR) predicate

    v16h a;
    if (blockFull || mrow < N) {
        const float* xp = x + (size_t)mrow * DIN + kbase;
        #pragma unroll
        for (int j = 0; j < 8; ++j) {
            a[j]     = (_Float16)xp[j];        // halves 0..7  -> K = kbase+j
            a[j + 8] = (_Float16)xp[j + 16];   // halves 8..15 -> K = kbase+16+j
        }
    } else {
        #pragma unroll
        for (int j = 0; j < 16; ++j) a[j] = (_Float16)0.f;
    }

    const int bbase  = (lane < 16) ? 0 : 8;    // B-frag: same K order as A (table layout)
    const int mstore = mbase + 8 * (lane >> 4);
    const int ncol   = lane & 15;

    if (blockFull) {
        // Fast path: no guards anywhere.
        #pragma unroll
        for (int nt = 0; nt < 8; ++nt) {
            const int n = nt * 16 + ncol;
            const _Float16* wt = WhT + n * DIN + bbase;
            v16h b;
            #pragma unroll
            for (int j = 0; j < 8; ++j) { b[j] = wt[j]; b[j + 8] = wt[j + 16]; }

            v8f c = {};
            c = __builtin_amdgcn_wmma_f32_16x16x32_f16(
                    false, a, false, b, (short)0, c, false, false);

            float* hp = h + (size_t)mstore * HC + n;
            #pragma unroll
            for (int r = 0; r < 8; ++r) hp[(size_t)r * HC] = c[r];
        }
    } else {
        // Trailing partial block only.
        #pragma unroll
        for (int nt = 0; nt < 8; ++nt) {
            const int n = nt * 16 + ncol;
            const _Float16* wt = WhT + n * DIN + bbase;
            v16h b;
            #pragma unroll
            for (int j = 0; j < 8; ++j) { b[j] = wt[j]; b[j + 8] = wt[j + 16]; }

            v8f c = {};
            c = __builtin_amdgcn_wmma_f32_16x16x32_f16(
                    false, a, false, b, (short)0, c, false, false);

            float* hp = h + (size_t)mstore * HC + n;
            #pragma unroll
            for (int r = 0; r < 8; ++r)
                if (mstore + r < N) hp[(size_t)r * HC] = c[r];
        }
    }
}

// ---------------- Kernel 2: per-(node,head) scores + re-init scratch
__global__ void gat_scores_init(
    const float* __restrict__ h, const float* __restrict__ att_src,
    const float* __restrict__ att_dst,
    float* __restrict__ a_s, float* __restrict__ a_d,
    float* __restrict__ amax, float* __restrict__ den,
    float* __restrict__ accum, int N)
{
    int t = blockIdx.x * blockDim.x + threadIdx.x;
    if (t >= N * NH) return;
    int nd = t >> 3, hh = t & 7;
    const float* hp = h + (size_t)nd * HC + hh * CC;
    const float* as = att_src + hh * CC;
    const float* ad = att_dst + hh * CC;
    float s0 = 0.f, s1 = 0.f;
    #pragma unroll
    for (int c = 0; c < CC; ++c) { float v = hp[c]; s0 += v * as[c]; s1 += v * ad[c]; }
    a_s[t] = s0;
    a_d[t] = s1;
    amax[t] = -__builtin_huge_valf();
    den[t]  = 0.f;
    float* ac = accum + (size_t)nd * HC + hh * CC;
    #pragma unroll
    for (int c = 0; c < CC; ++c) ac[c] = 0.f;
}

__device__ __forceinline__ void atomicMaxFloat(float* addr, float val) {
    // Sign-split trick: monotone on int for >=0, reverse-monotone on uint for <0.
    if (val >= 0.f) atomicMax((int*)addr, __float_as_int(val));
    else            atomicMin((unsigned int*)addr, __float_as_uint(val));
}

__device__ __forceinline__ float leaky02(float v) { return v >= 0.f ? v : 0.2f * v; }

// ---------------- Kernel 3: segment max of LeakyReLU(a_s[src]+a_d[dst]) over dst
__global__ void gat_edge_max(const int* __restrict__ adj, int E, int Etot,
    const float* __restrict__ a_s, const float* __restrict__ a_d,
    float* __restrict__ amax)
{
    int e = blockIdx.x * blockDim.x + threadIdx.x;
    if (e >= Etot) return;
    int s = (e < E) ? adj[e]     : (e - E);
    int d = (e < E) ? adj[E + e] : (e - E);
    const float* ps = a_s + (size_t)s * NH;
    const float* pd = a_d + (size_t)d * NH;
    float* pm = amax + (size_t)d * NH;
    #pragma unroll
    for (int hh = 0; hh < NH; ++hh)
        atomicMaxFloat(pm + hh, leaky02(ps[hh] + pd[hh]));
}

// ---------------- Kernel 4: denominator: sum exp(alpha - amax[dst])
__global__ void gat_edge_den(const int* __restrict__ adj, int E, int Etot,
    const float* __restrict__ a_s, const float* __restrict__ a_d,
    const float* __restrict__ amax, float* __restrict__ den)
{
    int e = blockIdx.x * blockDim.x + threadIdx.x;
    if (e >= Etot) return;
    int s = (e < E) ? adj[e]     : (e - E);
    int d = (e < E) ? adj[E + e] : (e - E);
    const float* ps = a_s + (size_t)s * NH;
    const float* pd = a_d + (size_t)d * NH;
    const float* pm = amax + (size_t)d * NH;
    float* pden = den + (size_t)d * NH;
    #pragma unroll
    for (int hh = 0; hh < NH; ++hh)
        atomicAdd(pden + hh, expf(leaky02(ps[hh] + pd[hh]) - pm[hh]));
}

// ---------------- Kernel 5: weighted scatter accum[dst] += w * h[src]
// One thread per (edge, head): 16 contiguous reads + 16 f32 atomics into L2.
__global__ void gat_edge_scatter(const int* __restrict__ adj, int E, long Tot,
    const float* __restrict__ a_s, const float* __restrict__ a_d,
    const float* __restrict__ amax, const float* __restrict__ den,
    const float* __restrict__ h, float* __restrict__ accum)
{
    long t = (long)blockIdx.x * blockDim.x + threadIdx.x;
    if (t >= Tot) return;
    int e  = (int)(t >> 3);
    int hh = (int)(t & 7);
    int s = (e < E) ? adj[e]     : (e - E);
    int d = (e < E) ? adj[E + e] : (e - E);
    float al = leaky02(a_s[(size_t)s * NH + hh] + a_d[(size_t)d * NH + hh]);
    float ex = expf(al - amax[(size_t)d * NH + hh]);
    float w  = ex / fmaxf(den[(size_t)d * NH + hh], 1e-16f);
    const float* hs = h + (size_t)s * HC + hh * CC;
    float* ap = accum + (size_t)d * HC + hh * CC;
    #pragma unroll
    for (int c = 0; c < CC; ++c) atomicAdd(ap + c, w * hs[c]);
}

// ---------------- Kernel 6: head mean + bias + leaky_relu(0.01)
__global__ void gat_finalize(const float* __restrict__ accum,
    const float* __restrict__ bias, float* __restrict__ out, int N)
{
    int t = blockIdx.x * blockDim.x + threadIdx.x;
    if (t >= N * CC) return;
    int nd = t >> 4, c = t & 15;
    const float* ap = accum + (size_t)nd * HC + c;
    float sum = 0.f;
    #pragma unroll
    for (int hh = 0; hh < NH; ++hh) sum += ap[hh * CC];
    float y = sum * (1.f / NH) + bias[c];
    out[t] = (y >= 0.f) ? y : 0.01f * y;
}

extern "C" void kernel_launch(void* const* d_in, const int* in_sizes, int n_in,
                              void* d_out, int out_size, void* d_ws, size_t ws_size,
                              hipStream_t stream)
{
    const int* adj = (const int*)d_in[0];
    const int  E    = in_sizes[0] / 2;
    const int  N    = in_sizes[1] / DIN;
    const int  Etot = E + N;

    float* ws    = (float*)d_ws;
    float* h     = ws;                          // N*128
    float* a_s   = h    + (size_t)N * HC;       // N*8
    float* a_d   = a_s  + (size_t)N * NH;       // N*8
    float* amax  = a_d  + (size_t)N * NH;       // N*8
    float* den   = amax + (size_t)N * NH;       // N*8
    float* accum = den  + (size_t)N * NH;       // N*128

    float* out = (float*)d_out;
    const long Tot = (long)Etot * NH;

    for (int f = 0; f < 3; ++f) {
        const float* x    = (const float*)d_in[1 + 5 * f];
        const float* W    = (const float*)d_in[2 + 5 * f];
        const float* atts = (const float*)d_in[3 + 5 * f];
        const float* attd = (const float*)d_in[4 + 5 * f];
        const float* bias = (const float*)d_in[5 + 5 * f];

        gat_gemm_wmma<<<(N + 127) / 128, 256, 0, stream>>>(x, W, h, N);
        gat_scores_init<<<(N * NH + 255) / 256, 256, 0, stream>>>(
            h, atts, attd, a_s, a_d, amax, den, accum, N);
        gat_edge_max<<<(Etot + 255) / 256, 256, 0, stream>>>(adj, E, Etot, a_s, a_d, amax);
        gat_edge_den<<<(Etot + 255) / 256, 256, 0, stream>>>(adj, E, Etot, a_s, a_d, amax, den);
        gat_edge_scatter<<<(unsigned)((Tot + 255) / 256), 256, 0, stream>>>(
            adj, E, Tot, a_s, a_d, amax, den, h, accum);
        gat_finalize<<<(N * CC + 255) / 256, 256, 0, stream>>>(
            accum, bias, out + (size_t)f * N * CC, N);
    }
}
